// Attention_73899207295255
// MI455X (gfx1250) — compile-verified, matrix operand order checked by reading, not run on hardware
//
#include <hip/hip_runtime.h>
#include <hip/hip_bf16.h>
#include <math.h>

// ---------------------------------------------------------------------------
// DoReFa-8bit quantized GPT2 attention block for gfx1250 (MI455X).
// f16 WMMA (v_wmma_f32_16x16x32_f16) for all three matmul stages.
// Async global->LDS staging (ASYNCcnt) for k-tiles / proj A-tiles.
// B=8 S=1024 NX=768 H=12 hd=64
// ---------------------------------------------------------------------------

typedef __attribute__((ext_vector_type(16))) _Float16 v16h;
typedef __attribute__((ext_vector_type(8)))  float    v8f;

#define Bsz 8
#define Ssz 1024
#define NX  768
#define NH  12
#define HD  64
#define N3  2304
#define MROWS (Bsz*Ssz)              // 8192
#define AOFF  ((long long)MROWS*NX)  // 6291456, start of present in d_out
#define PRES  ((long long)Bsz*NH*Ssz*HD) // 6291456 per tensor

// ---- CDNA5 helpers --------------------------------------------------------

__device__ __forceinline__ v8f wmma_f16(v16h a, v16h b, v8f c) {
  return __builtin_amdgcn_wmma_f32_16x16x32_f16(
      /*neg_a=*/false, a, /*neg_b=*/false, b,
      /*c_mod=*/(short)0, c, /*reuse_a=*/false, /*reuse_b=*/false);
}

// Async DMA 16 bytes/lane: global -> LDS, tracked by ASYNCcnt.
// Generic LDS pointer: low 32 bits == wave-relative LDS byte offset.
__device__ __forceinline__ void async_b128(const _Float16* lds_dst,
                                           const _Float16* g_src) {
  unsigned l = (unsigned)(unsigned long long)(uintptr_t)lds_dst;
  unsigned long long g = (unsigned long long)(uintptr_t)g_src;
  asm volatile("global_load_async_to_lds_b128 %0, %1, off"
               :: "v"(l), "v"(g) : "memory");
}

// Load a 16x32 f16 A/B fragment. `p` points at element (row0, k0) of a
// row-major (A: M-major, B: N-major) tile with leading dim `ld` (halfs, even).
// Per ISA 16-bit A layout: lane lo=lane&15 -> row, hi=lane>>4 -> K +8/+24;
// half pair v (0..7): K = (v>>2)*16 + hi*8 + (v&3)*2
__device__ __forceinline__ v16h frag_load16(const _Float16* p, int ld) {
  const int lane = threadIdx.x & 31;
  const int lo = lane & 15, hi = lane >> 4;
  union { v16h h; unsigned u[8]; } r;
#pragma unroll
  for (int v = 0; v < 8; ++v) {
    const int kb = ((v >> 2) << 4) + (hi << 3) + ((v & 3) << 1);
    r.u[v] = *(const unsigned*)(p + lo * ld + kb);
  }
  return r.h;
}

// ---- weight_quant support -------------------------------------------------

__global__ void init_mx_kernel(unsigned* mx) {
  if (threadIdx.x < 2) mx[threadIdx.x] = 0u;
}

__global__ __launch_bounds__(256) void maxabs_tanh_kernel(
    const float* __restrict__ w, int n, unsigned* __restrict__ out) {
  float m = 0.f;
  for (int i = blockIdx.x * blockDim.x + threadIdx.x; i < n;
       i += gridDim.x * blockDim.x)
    m = fmaxf(m, fabsf(tanhf(w[i])));
#pragma unroll
  for (int s = 1; s < 32; s <<= 1)
    m = fmaxf(m, __shfl_xor(m, s, 32));
  __shared__ float red[8];
  if ((threadIdx.x & 31) == 0) red[threadIdx.x >> 5] = m;
  __syncthreads();
  if (threadIdx.x == 0) {
    float mm = red[0];
#pragma unroll
    for (int i = 1; i < 8; ++i) mm = fmaxf(mm, red[i]);
    atomicMax(out, __float_as_uint(mm));  // non-neg floats: bit order == order
  }
}

__global__ __launch_bounds__(256) void quant_weights_kernel(
    const float* __restrict__ w, _Float16* __restrict__ wq, int n,
    const unsigned* __restrict__ mx) {
  int i = blockIdx.x * blockDim.x + threadIdx.x;
  if (i >= n) return;
  const float M = __uint_as_float(*mx);
  float t = tanhf(w[i]);
  float tq = t / (2.f * M) + 0.5f;
  float q = rintf(tq * 255.f) * (1.f / 255.f);   // RNE == jnp.round
  wq[i] = (_Float16)(2.f * q - 1.f);
}

// ---- QKV GEMM + bias + act_quant + head scatter ---------------------------
// C[8192 x 2304] = X[8192 x 768] * Wq[768 x 2304]
// tile 128x128x32, 8 waves (4 M x 2 N), wave tile 32x64 = 2x4 fragments.

__global__ __launch_bounds__(256) void gemm_qkv_kernel(
    const float* __restrict__ X, const _Float16* __restrict__ W,
    const float* __restrict__ bias, _Float16* __restrict__ Qo,
    _Float16* __restrict__ Ko, _Float16* __restrict__ Vo,
    float* __restrict__ out) {
  __shared__ __align__(16) _Float16 At[128 * 40];
  __shared__ __align__(16) _Float16 Bt[128 * 40];
  const int tid = threadIdx.x;
  const int wid = tid >> 5;
  const int lane = tid & 31;
  const int lo = lane & 15, hi = lane >> 4;
  const int wm = wid >> 1, wn = wid & 1;
  const int m0 = blockIdx.y << 7;
  const int n0 = blockIdx.x << 7;

  const v8f zero = {0.f,0.f,0.f,0.f,0.f,0.f,0.f,0.f};
  v8f acc[2][4];
#pragma unroll
  for (int im = 0; im < 2; ++im)
#pragma unroll
    for (int in = 0; in < 4; ++in) acc[im][in] = zero;

  for (int k0 = 0; k0 < NX; k0 += 32) {
    __syncthreads();
    for (int i = tid; i < 128 * 32; i += 256) {       // A: f32 -> f16, M-major
      int r = i >> 5, c = i & 31;
      At[r * 40 + c] = (_Float16)X[(long long)(m0 + r) * NX + (k0 + c)];
    }
    for (int i = tid; i < 32 * 128; i += 256) {       // B: transpose to N-major
      int r = i >> 7, c = i & 127;
      Bt[c * 40 + r] = W[(long long)(k0 + r) * N3 + (n0 + c)];
    }
    if (k0 + 32 < NX) {                               // hint next k-slice
      __builtin_prefetch(&X[(long long)(m0 + (tid >> 5)) * NX + (k0 + 32) + (tid & 31)], 0, 0);
      __builtin_prefetch(&W[(long long)(k0 + 32 + (tid >> 7)) * N3 + n0 + (tid & 127)], 0, 0);
    }
    __syncthreads();
    v16h af0 = frag_load16(&At[(wm * 32) * 40], 40);
    v16h af1 = frag_load16(&At[(wm * 32 + 16) * 40], 40);
#pragma unroll
    for (int in = 0; in < 4; ++in) {
      v16h bf = frag_load16(&Bt[(wn * 64 + in * 16) * 40], 40);
      acc[0][in] = wmma_f16(af0, bf, acc[0][in]);
      acc[1][in] = wmma_f16(af1, bf, acc[1][in]);
    }
  }

  // epilogue: + bias, act_quant, scatter to [B,H,S,hd]; k/v also -> present
#pragma unroll
  for (int im = 0; im < 2; ++im)
#pragma unroll
    for (int in = 0; in < 4; ++in)
#pragma unroll
      for (int r = 0; r < 8; ++r) {
        int m = m0 + wm * 32 + im * 16 + r + (hi << 3);
        int f = n0 + wn * 64 + in * 16 + lo;
        float val = acc[im][in][r] + bias[f];
        float qv = fminf(fmaxf(val, 0.f), 1.f);
        qv = rintf(qv * 255.f) * (1.f / 255.f);
        int bidx = m >> 10, s = m & 1023;
        int sec = f / NX, w = f % NX;
        int hh = w >> 6, d = w & 63;
        long long hidx = (((long long)(bidx * NH + hh)) * Ssz + s) * HD + d;
        if (sec == 0) {
          Qo[hidx] = (_Float16)qv;
        } else if (sec == 1) {
          Ko[hidx] = (_Float16)qv;
          out[AOFF + hidx] = qv;               // present[0] = k
        } else {
          Vo[hidx] = (_Float16)qv;
          out[AOFF + PRES + hidx] = qv;        // present[1] = v
        }
      }
}

// ---- flash-style causal attention -----------------------------------------
// grid: (S/128, B*H). 8 waves x 16 q rows. Double-buffered 32-key tiles:
// k via async global->LDS (ASYNCcnt), v via register prefetch + LDS transpose.

__global__ __launch_bounds__(256) void attn_kernel(
    const _Float16* __restrict__ Q, const _Float16* __restrict__ K,
    const _Float16* __restrict__ V, _Float16* __restrict__ A) {
  __shared__ __align__(16) _Float16 kt[2][32 * 72];   // [key][hd]
  __shared__ __align__(16) _Float16 vt[2][64 * 40];   // [hd][key] (transposed)
  __shared__ __align__(16) _Float16 pbuf[8 * 16 * 36];

  const int tid = threadIdx.x;
  const int wid = tid >> 5;
  const int lane = tid & 31;
  const int lo = lane & 15, hi = lane >> 4;

  const int bh = blockIdx.y;            // 0..95
  const int b = bh / NH, h = bh % NH;
  const int qb = blockIdx.x << 7;
  const int qrow0 = qb + (wid << 4);

  const long long base = (long long)bh * Ssz * HD;
  const _Float16* qptr = Q + base + (long long)qrow0 * HD;
  const v16h qf0 = frag_load16(qptr, HD);            // hd 0..31
  const v16h qf1 = frag_load16(qptr + 32, HD);       // hd 32..63

  const v8f zero = {0.f,0.f,0.f,0.f,0.f,0.f,0.f,0.f};
  v8f acco[4];
#pragma unroll
  for (int t = 0; t < 4; ++t) acco[t] = zero;
  float mrow[8], lrow[8];
#pragma unroll
  for (int r = 0; r < 8; ++r) { mrow[r] = -1e30f; lrow[r] = 0.f; }

  // staging coords: 256 threads x 16B = one 32x64 f16 tile
  const int key = tid >> 3, dg = (tid & 7) << 3;
  const int nblk = (qb >> 5) + 4;       // #32-key tiles (causal bound)

  // pipeline prologue: tile 0
  uint4 vcur = *(const uint4*)(V + base + (long long)key * HD + dg);
  async_b128(&kt[0][key * 72 + dg], K + base + (long long)key * HD + dg);
  uint4 vnext = vcur;

  for (int it = 0; it < nblk; ++it) {
    const int kb = it << 5;
    const int buf = it & 1;

    // transpose-store prefetched v tile
    { union { uint4 u; _Float16 hh[8]; } c; c.u = vcur;
#pragma unroll
      for (int j = 0; j < 8; ++j) vt[buf][(dg + j) * 40 + key] = c.hh[j];
    }
    if (it + 1 < nblk) {                // prefetch tile it+1
      const long long nb = base + (long long)(kb + 32 + key) * HD + dg;
      vnext = *(const uint4*)(V + nb);
      async_b128(&kt[buf ^ 1][key * 72 + dg], K + nb);
      asm volatile("s_wait_asynccnt 0x1" ::: "memory");  // tile `it` landed
    } else {
      asm volatile("s_wait_asynccnt 0x0" ::: "memory");
    }
    __syncthreads();

    float pl[2][8], mnew[8], scl[8];
#pragma unroll
    for (int nh = 0; nh < 2; ++nh) {
      v8f s = zero;
      v16h b0 = frag_load16(&kt[buf][(nh * 16) * 72 + 0], 72);
      v16h b1 = frag_load16(&kt[buf][(nh * 16) * 72 + 32], 72);
      s = wmma_f16(qf0, b0, s);
      s = wmma_f16(qf1, b1, s);
#pragma unroll
      for (int r = 0; r < 8; ++r) {
        int mg = qrow0 + r + (hi << 3);
        int ng = kb + nh * 16 + lo;
        float sv = s[r] * 0.125f;                    // 1/sqrt(64)
        pl[nh][r] = (ng <= mg) ? sv : -10000.0f;     // causal mask (ref: -1e4)
      }
    }
#pragma unroll
    for (int r = 0; r < 8; ++r) {
      float rm = fmaxf(pl[0][r], pl[1][r]);
#pragma unroll
      for (int mk = 1; mk <= 8; mk <<= 1)            // within 16-lane group
        rm = fmaxf(rm, __shfl_xor(rm, mk, 32));
      mnew[r] = fmaxf(mrow[r], rm);
      scl[r] = __expf(mrow[r] - mnew[r]);
      float p0 = __expf(pl[0][r] - mnew[r]);
      float p1 = __expf(pl[1][r] - mnew[r]);
      pl[0][r] = p0; pl[1][r] = p1;
      float rs = p0 + p1;
#pragma unroll
      for (int mk = 1; mk <= 8; mk <<= 1)
        rs += __shfl_xor(rs, mk, 32);
      lrow[r] = lrow[r] * scl[r] + rs;
      mrow[r] = mnew[r];
    }
#pragma unroll
    for (int t = 0; t < 4; ++t)
#pragma unroll
      for (int r = 0; r < 8; ++r) acco[t][r] *= scl[r];

    // D-layout p -> per-wave LDS tile [16 rows][32 keys] -> A fragment
    _Float16* pb = &pbuf[wid * 16 * 36];
#pragma unroll
    for (int nh = 0; nh < 2; ++nh)
#pragma unroll
      for (int r = 0; r < 8; ++r)
        pb[(r + (hi << 3)) * 36 + nh * 16 + lo] = (_Float16)pl[nh][r];
    asm volatile("s_wait_dscnt 0" ::: "memory");     // own-wave DS writes visible
    v16h pf = frag_load16(pb, 36);
#pragma unroll
    for (int t = 0; t < 4; ++t) {
      v16h bv = frag_load16(&vt[buf][(t * 16) * 40], 40);
      acco[t] = wmma_f16(pf, bv, acco[t]);
    }
    __syncthreads();                                 // buf reusable next round
    vcur = vnext;
  }

  // normalize and write merged heads: A[b*1024+s][h*64+d] (f16)
#pragma unroll
  for (int t = 0; t < 4; ++t)
#pragma unroll
    for (int r = 0; r < 8; ++r) {
      int s = qrow0 + r + (hi << 3);
      int col = h * HD + t * 16 + lo;
      float o = acco[t][r] / lrow[r];
      A[((long long)b * Ssz + s) * NX + col] = (_Float16)o;
    }
}

// ---- output projection GEMM -----------------------------------------------
// out[8192 x 768] = Am[8192 x 768] * Wq_proj[768 x 768] + b_proj
// A tile staged by async global->LDS (f16 row-major, no conversion needed).

__global__ __launch_bounds__(256) void gemm_proj_kernel(
    const _Float16* __restrict__ Am, const _Float16* __restrict__ W,
    const float* __restrict__ bias, float* __restrict__ out) {
  __shared__ __align__(16) _Float16 At[128 * 40];
  __shared__ __align__(16) _Float16 Bt[128 * 40];
  const int tid = threadIdx.x;
  const int wid = tid >> 5;
  const int lane = tid & 31;
  const int lo = lane & 15, hi = lane >> 4;
  const int wm = wid >> 1, wn = wid & 1;
  const int m0 = blockIdx.y << 7;
  const int n0 = blockIdx.x << 7;

  const v8f zero = {0.f,0.f,0.f,0.f,0.f,0.f,0.f,0.f};
  v8f acc[2][4];
#pragma unroll
  for (int im = 0; im < 2; ++im)
#pragma unroll
    for (int in = 0; in < 4; ++in) acc[im][in] = zero;

  for (int k0 = 0; k0 < NX; k0 += 32) {
    __syncthreads();
    // A tile: 512 x 16B chunks via async DMA (2 per thread)
#pragma unroll
    for (int cc = 0; cc < 2; ++cc) {
      int c = tid + (cc << 8);
      int r = c >> 2, d = (c & 3) << 3;
      async_b128(&At[r * 40 + d], &Am[(long long)(m0 + r) * NX + k0 + d]);
    }
    for (int i = tid; i < 32 * 128; i += 256) {       // B: transpose to N-major
      int r = i >> 7, c = i & 127;
      Bt[c * 40 + r] = W[(long long)(k0 + r) * NX + (n0 + c)];
    }
    if (k0 + 32 < NX)
      __builtin_prefetch(&W[(long long)(k0 + 32 + (tid >> 7)) * NX + n0 + (tid & 127)], 0, 0);
    asm volatile("s_wait_asynccnt 0x0" ::: "memory");
    __syncthreads();
    v16h af0 = frag_load16(&At[(wm * 32) * 40], 40);
    v16h af1 = frag_load16(&At[(wm * 32 + 16) * 40], 40);
#pragma unroll
    for (int in = 0; in < 4; ++in) {
      v16h bf = frag_load16(&Bt[(wn * 64 + in * 16) * 40], 40);
      acc[0][in] = wmma_f16(af0, bf, acc[0][in]);
      acc[1][in] = wmma_f16(af1, bf, acc[1][in]);
    }
  }
#pragma unroll
  for (int im = 0; im < 2; ++im)
#pragma unroll
    for (int in = 0; in < 4; ++in)
#pragma unroll
      for (int r = 0; r < 8; ++r) {
        int m = m0 + wm * 32 + im * 16 + r + (hi << 3);
        int f = n0 + wn * 64 + in * 16 + lo;
        out[(long long)m * NX + f] = acc[im][in][r] + bias[f];
      }
}

// ---- Host side ------------------------------------------------------------

extern "C" void kernel_launch(void* const* d_in, const int* in_sizes, int n_in,
                              void* d_out, int out_size, void* d_ws, size_t ws_size,
                              hipStream_t stream) {
  const float* x      = (const float*)d_in[0];   // [8,1024,768]
  const float* W_attn = (const float*)d_in[1];   // [768,2304]
  const float* b_attn = (const float*)d_in[2];   // [2304]
  const float* W_proj = (const float*)d_in[3];   // [768,768]
  const float* b_proj = (const float*)d_in[4];   // [768]
  float* out = (float*)d_out;                    // a(6291456) ++ present(12582912)

  // workspace carve-up (256B aligned)
  char* ws = (char*)d_ws;
  size_t off = 0;
  auto carve = [&](size_t bytes) {
    size_t o = off;
    off = (off + bytes + 255) & ~(size_t)255;
    return o;
  };
  unsigned*  mx  = (unsigned*)(ws + carve(2 * sizeof(unsigned)));
  _Float16*  wqa = (_Float16*)(ws + carve((size_t)NX * N3 * 2));
  _Float16*  wqp = (_Float16*)(ws + carve((size_t)NX * NX * 2));
  _Float16*  qh  = (_Float16*)(ws + carve((size_t)PRES * 2));
  _Float16*  kh  = (_Float16*)(ws + carve((size_t)PRES * 2));
  _Float16*  vh  = (_Float16*)(ws + carve((size_t)PRES * 2));
  _Float16*  am  = (_Float16*)(ws + carve((size_t)MROWS * NX * 2));
  (void)ws_size; (void)in_sizes; (void)n_in; (void)out_size;

  const int nWa = NX * N3;   // 1769472
  const int nWp = NX * NX;   // 589824

  init_mx_kernel<<<1, 32, 0, stream>>>(mx);
  maxabs_tanh_kernel<<<512, 256, 0, stream>>>(W_attn, nWa, mx + 0);
  maxabs_tanh_kernel<<<512, 256, 0, stream>>>(W_proj, nWp, mx + 1);
  quant_weights_kernel<<<(nWa + 255) / 256, 256, 0, stream>>>(W_attn, wqa, nWa, mx + 0);
  quant_weights_kernel<<<(nWp + 255) / 256, 256, 0, stream>>>(W_proj, wqp, nWp, mx + 1);

  gemm_qkv_kernel<<<dim3(N3 / 128, MROWS / 128), 256, 0, stream>>>(
      x, wqa, b_attn, qh, kh, vh, out);

  attn_kernel<<<dim3(Ssz / 128, Bsz * NH), 256, 0, stream>>>(qh, kh, vh, am);

  gemm_proj_kernel<<<dim3(NX / 128, MROWS / 128), 256, 0, stream>>>(
      am, wqp, b_proj, out);
}
